// GCN_50757923504498
// MI455X (gfx1250) — matrix operand order repaired and checked by reading, not compiled, not run on hardware
//
#include <hip/hip_runtime.h>
#include <math.h>

typedef __attribute__((ext_vector_type(16))) _Float16 v16h;
typedef __attribute__((ext_vector_type(8)))  float    v8f;

// ---------------- problem constants ----------------
constexpr int BATCH   = 64;
constexpr int NNODES  = 10000;
constexpr int FEAT    = 16;
constexpr int NEDGE   = 160000;
constexpr int KTOP    = 1000;
constexpr int FIRSTIN = KTOP * FEAT;   // 16000
constexpr int HID1    = 512;
constexpr int HID2    = 128;
constexpr float BN_EPS = 1e-5f;
constexpr int KCHUNK1 = 2000;          // split-K chunk for GEMM1 (8 chunks)

// ---------------- workspace layout (bytes, 256-aligned) ----------------
constexpr size_t OFF_AGG    = 0;            // f32 [B][N][16]   40,960,000
constexpr size_t OFF_H      = 40960000;     // f32 [B][N][16]   40,960,000
constexpr size_t OFF_SCORE  = 81920000;     // f32 [B][N]        2,560,000
constexpr size_t OFF_FLAT   = 84480000;     // f16 [B][16000]    2,048,000
constexpr size_t OFF_W1H    = 86528000;     // f16 [512][16000] 16,384,000
constexpr size_t OFF_W2H    = 102912000;    // f16 [128][512]      131,072
constexpr size_t OFF_Y1     = 103043072;    // f32 [64][512]       131,072
constexpr size_t OFF_H1H    = 103174144;    // f16 [64][512]        65,536
constexpr size_t OFF_Y2     = 103239680;    // f32 [64][128]        32,768
constexpr size_t OFF_H2H    = 103272448;    // f16 [64][128]        16,384
constexpr size_t OFF_DEG    = 103305216;    // i32 [10000]          40,192 (padded)
constexpr size_t OFF_ROWPTR = 103345408;    // i32 [10001]          40,192 (padded)
constexpr size_t OFF_CURSOR = 103385600;    // i32 [10000]          40,192 (padded)
constexpr size_t OFF_CSRC   = 103425792;    // i32 [160000]        640,000

// ---------------- utility kernels ----------------
__global__ void zero_b32(unsigned* __restrict__ p, int n) {
    int i = blockIdx.x * blockDim.x + threadIdx.x;
    if (i < n) p[i] = 0u;
}

__global__ void cvt_f32_f16(const float* __restrict__ in, _Float16* __restrict__ out, int n) {
    int i = blockIdx.x * blockDim.x + threadIdx.x;
    if (i < n) out[i] = (_Float16)in[i];
}

// ---------------- CSR-by-dst construction (shared by all 64 graphs) ----------------
__global__ void hist_dst(const int* __restrict__ dst, int* __restrict__ deg) {
    int e = blockIdx.x * blockDim.x + threadIdx.x;
    if (e < NEDGE) atomicAdd(&deg[dst[e]], 1);
}

// one block, 1024 threads: exclusive prefix over 10000 degrees -> rowptr, cursor
__global__ void scan_rowptr(const int* __restrict__ deg, int* __restrict__ rowptr,
                            int* __restrict__ cursor) {
    __shared__ int part[1024];
    const int CH = 10;                          // 1024*10 >= 10000
    int t = threadIdx.x;
    int base = t * CH;
    int s = 0;
    for (int i = 0; i < CH; ++i) { int idx = base + i; if (idx < NNODES) s += deg[idx]; }
    part[t] = s;
    __syncthreads();
    for (int off = 1; off < 1024; off <<= 1) {  // Hillis-Steele inclusive scan
        int v = (t >= off) ? part[t - off] : 0;
        __syncthreads();
        part[t] += v;
        __syncthreads();
    }
    int run = (t == 0) ? 0 : part[t - 1];
    for (int i = 0; i < CH; ++i) {
        int idx = base + i;
        if (idx < NNODES) { rowptr[idx] = run; cursor[idx] = run; run += deg[idx]; }
    }
    if (t == 1023) rowptr[NNODES] = part[1023];
}

__global__ void fill_csr(const int* __restrict__ src, const int* __restrict__ dst,
                         int* __restrict__ cursor, int* __restrict__ csrc) {
    int e = blockIdx.x * blockDim.x + threadIdx.x;
    if (e < NEDGE) {
        int p = atomicAdd(&cursor[dst[e]], 1);
        csrc[p] = src[e];
    }
}

// ---------------- atomic-free aggregation: agg[b][n] = sum_{e: dst=n} xin[b][src_e] -------
__global__ void gather_agg(const float* __restrict__ xin, const int* __restrict__ rowptr,
                           const int* __restrict__ csrc, float* __restrict__ agg) {
    int idx = blockIdx.x * blockDim.x + threadIdx.x;
    if (idx >= BATCH * NNODES) return;
    int b = idx / NNODES, n = idx % NNODES;
    int p0 = rowptr[n], p1 = rowptr[n + 1];
    float a[FEAT];
    #pragma unroll
    for (int f = 0; f < FEAT; ++f) a[f] = 0.0f;
    const float* xb = xin + (size_t)b * NNODES * FEAT;
    for (int p = p0; p < p1; ++p) {
        const float4* xr = (const float4*)(xb + (size_t)csrc[p] * FEAT);
        float4 v0 = xr[0], v1 = xr[1], v2 = xr[2], v3 = xr[3];
        a[0]+=v0.x; a[1]+=v0.y; a[2]+=v0.z; a[3]+=v0.w;
        a[4]+=v1.x; a[5]+=v1.y; a[6]+=v1.z; a[7]+=v1.w;
        a[8]+=v2.x; a[9]+=v2.y; a[10]+=v2.z; a[11]+=v2.w;
        a[12]+=v3.x; a[13]+=v3.y; a[14]+=v3.z; a[15]+=v3.w;
    }
    float4* ar = (float4*)(agg + (size_t)idx * FEAT);
    ar[0] = make_float4(a[0], a[1], a[2], a[3]);
    ar[1] = make_float4(a[4], a[5], a[6], a[7]);
    ar[2] = make_float4(a[8], a[9], a[10], a[11]);
    ar[3] = make_float4(a[12], a[13], a[14], a[15]);
}

// ---------------- GraphConv1 via WMMA ----------------
// h = relu( agg @ Wrel.T + x @ Wroot.T + brel )
// A(16x32) = [agg_tile | x_tile], B(32x16): rows 0..15 = Wrel.T, 16..31 = Wroot.T
__global__ void conv1_wmma(const float* __restrict__ x, const float* __restrict__ agg,
                           const float* __restrict__ Wrel, const float* __restrict__ Wroot,
                           const float* __restrict__ brel, float* __restrict__ hout) {
    int wid  = blockIdx.x;                 // one wave (32 threads) per block
    int lane = threadIdx.x & 31;
    int b    = wid / (NNODES / 16);
    int n0   = (wid % (NNODES / 16)) * 16;
    int q    = lane & 15;
    bool hi  = lane >= 16;

    // B fragment: lane l, elem h<8 -> K=lo+h (Wrel), h>=8 -> K=16+lo+(h-8) (Wroot), N=q
    v16h bf;
    #pragma unroll
    for (int h = 0; h < 16; ++h) bf[h] = (_Float16)(h < 8 ? Wrel : Wroot)[q * 16 + (hi ? 8 : 0) + (h & 7)];

    // A fragment: lane holds row m=q; halves 0..7: K=lo..lo+7 (agg),
    //             halves 8..15: K=16+lo..16+lo+7 (x), lo = hi?8:0
    int lo = hi ? 8 : 0;
    const float* arow = agg + ((size_t)b * NNODES + n0 + q) * FEAT + lo;
    const float* xrow = x   + ((size_t)b * NNODES + n0 + q) * FEAT + lo;
    v16h af;
    #pragma unroll
    for (int i = 0; i < 8; ++i) { af[i] = (_Float16)arow[i]; af[8 + i] = (_Float16)xrow[i]; }

    v8f c = {};
    c = __builtin_amdgcn_wmma_f32_16x16x32_f16(false, af, false, bf, (short)0, c, false, false);

    float bb = brel[q];                          // output feature N = q
    #pragma unroll
    for (int j = 0; j < 8; ++j) {
        int M = (hi ? 8 : 0) + j;                // node row within tile
        float v = c[j] + bb;
        hout[((size_t)b * NNODES + n0 + M) * FEAT + q] = v > 0.0f ? v : 0.0f;
    }
}

// ---------------- scorer: score = tanh(aggh.Wp_rel + bp + h.Wp_root) ----------------
__global__ void scorer(const float* __restrict__ aggh, const float* __restrict__ h,
                       const float* __restrict__ Wp_rel, const float* __restrict__ bp,
                       const float* __restrict__ Wp_root, float* __restrict__ score) {
    int i = blockIdx.x * blockDim.x + threadIdx.x;
    if (i >= BATCH * NNODES) return;
    const float* ar = aggh + (size_t)i * FEAT;
    const float* hr = h    + (size_t)i * FEAT;
    float s = bp[0];
    #pragma unroll
    for (int f = 0; f < FEAT; ++f) s += ar[f] * Wp_rel[f] + hr[f] * Wp_root[f];
    score[i] = tanhf(s);
}

// ---------------- per-graph sorted top-K (bitonic, 128KB dyn LDS) + gather ----------------
__global__ void topk_gather(const float* __restrict__ score, const float* __restrict__ h,
                            _Float16* __restrict__ flat) {
    extern __shared__ unsigned long long skey[];     // 16384 entries (128 KB, CDNA5 WGP LDS)
    const int M = 16384;
    int b = blockIdx.x, tid = threadIdx.x, nt = blockDim.x;

    for (int i = tid; i < M; i += nt) {
        float v = (i < NNODES) ? score[(size_t)b * NNODES + i] : -INFINITY;
        unsigned u = __float_as_uint(v);
        u = (u & 0x80000000u) ? ~u : (u | 0x80000000u);          // monotone f32->u32
        skey[i] = ((unsigned long long)u << 32) | (unsigned)(0xFFFFFFFFu - (unsigned)i);
    }
    __syncthreads();

    for (int k = 2; k <= M; k <<= 1) {
        for (int j = k >> 1; j > 0; j >>= 1) {
            for (int i = tid; i < M; i += nt) {
                int p = i ^ j;
                if (p > i) {
                    unsigned long long a = skey[i], c = skey[p];
                    bool sw = ((i & k) == 0) ? (a < c) : (a > c);  // descending
                    if (sw) { skey[i] = c; skey[p] = a; }
                }
            }
            __syncthreads();
        }
    }

    for (int p = tid; p < KTOP; p += nt) {
        unsigned long long kk = skey[p];
        unsigned u = (unsigned)(kk >> 32);
        u = (u & 0x80000000u) ? (u & 0x7FFFFFFFu) : ~u;           // inverse map
        float sv = __uint_as_float(u);
        int id = (int)(0xFFFFFFFFu - (unsigned)(kk & 0xFFFFFFFFu));
        const float* hr = h + ((size_t)b * NNODES + id) * FEAT;
        _Float16* fr = flat + (size_t)b * FIRSTIN + (size_t)p * FEAT;
        #pragma unroll
        for (int f = 0; f < FEAT; ++f) fr[f] = (_Float16)(hr[f] * sv);
    }
}

// ------- split-K f16 WMMA GEMM: C[64][N] += A[64][K] @ W[N][K].T (+bias from chunk 0) -----
// grid = (64/16) * (N/16) * (K/Klen) waves; C must be zeroed first.
__global__ void gemm_wmma_splitk(const _Float16* __restrict__ A, const _Float16* __restrict__ W,
                                 const float* __restrict__ bias, float* __restrict__ C,
                                 int Nout, int Klen) {
    int ntN     = Nout / 16;
    int tilesMN = (BATCH / 16) * ntN;
    int wid     = blockIdx.x;
    int kc      = wid / tilesMN;
    int rem     = wid % tilesMN;
    int mt      = rem / ntN, ntile = rem % ntN;
    int lane    = threadIdx.x & 31;
    int q       = lane & 15;
    bool hi     = lane >= 16;
    int Kdim    = Klen * (gridDim.x / tilesMN);
    int k0      = kc * Klen;

    const _Float16* ab = A + (size_t)(mt * 16 + q) * Kdim + (hi ? 8 : 0);
    const _Float16* wb = W + (size_t)(ntile * 16 + q) * Kdim + (hi ? 8 : 0);

    v8f c = {};
    // Klen is a multiple of 128: unroll x4 so 16 loads are in flight per wait,
    // letting WMMA co-execute with the load tail (MLP per wave, not just per grid).
    #pragma unroll 4
    for (int kb = k0; kb < k0 + Klen; kb += 32) {
        v16h af, bf;
        const _Float16* ap = ab + kb;
        #pragma unroll
        for (int i = 0; i < 8; ++i) { af[i] = ap[i]; af[8 + i] = ap[16 + i]; }
        const _Float16* wp = wb + kb;
        #pragma unroll
        for (int i = 0; i < 8; ++i) { bf[i] = wp[i]; bf[8 + i] = wp[16 + i]; }
        c = __builtin_amdgcn_wmma_f32_16x16x32_f16(false, af, false, bf, (short)0, c, false, false);
    }
    float bb = (kc == 0) ? bias[ntile * 16 + q] : 0.0f;
    #pragma unroll
    for (int j = 0; j < 8; ++j) {
        int mr = mt * 16 + (hi ? 8 : 0) + j;
        atomicAdd(&C[(size_t)mr * Nout + ntile * 16 + q], c[j] + bb);
    }
}

// ---------------- BatchNorm(train, biased var) over batch dim + optional ReLU, emit f16 ----
__global__ void bn_relu_f16(const float* __restrict__ y, const float* __restrict__ g,
                            const float* __restrict__ be, _Float16* __restrict__ outh,
                            int F, int dorelu) {
    int f = blockIdx.x * blockDim.x + threadIdx.x;
    if (f >= F) return;
    float m = 0.f;
    for (int r = 0; r < BATCH; ++r) m += y[(size_t)r * F + f];
    m /= BATCH;
    float v = 0.f;
    for (int r = 0; r < BATCH; ++r) { float d = y[(size_t)r * F + f] - m; v += d * d; }
    v /= BATCH;
    float is = rsqrtf(v + BN_EPS);
    for (int r = 0; r < BATCH; ++r) {
        float o = g[f] * (y[(size_t)r * F + f] - m) * is + be[f];
        if (dorelu && o < 0.f) o = 0.f;
        outh[(size_t)r * F + f] = (_Float16)o;
    }
}

// ---------------- head: y = h2 @ W3.T + b3 ; out = BN(y) ----------------
__global__ void head(const _Float16* __restrict__ h2, const float* __restrict__ W3,
                     const float* __restrict__ b3, const float* __restrict__ g3,
                     const float* __restrict__ be3, float* __restrict__ out) {
    __shared__ float yv[BATCH];
    int r = threadIdx.x;                      // 64 threads
    float s = b3[0];
    for (int f = 0; f < HID2; ++f) s += (float)h2[r * HID2 + f] * W3[f];
    yv[r] = s;
    __syncthreads();
    float m = 0.f;
    for (int i = 0; i < BATCH; ++i) m += yv[i];
    m /= BATCH;
    float v = 0.f;
    for (int i = 0; i < BATCH; ++i) { float d = yv[i] - m; v += d * d; }
    v /= BATCH;
    out[r] = g3[0] * (yv[r] - m) * rsqrtf(v + BN_EPS) + be3[0];
}

// ---------------- launcher ----------------
extern "C" void kernel_launch(void* const* d_in, const int* in_sizes, int n_in,
                              void* d_out, int out_size, void* d_ws, size_t ws_size,
                              hipStream_t stream) {
    const float* data    = (const float*)d_in[0];
    const int*   eidx    = (const int*)d_in[1];
    const float* W_rel1  = (const float*)d_in[2];
    const float* b_rel1  = (const float*)d_in[3];
    const float* W_root1 = (const float*)d_in[4];
    const float* Wp_rel  = (const float*)d_in[5];
    const float* bp_rel  = (const float*)d_in[6];
    const float* Wp_root = (const float*)d_in[7];
    const float* W1 = (const float*)d_in[8];  const float* b1 = (const float*)d_in[9];
    const float* g1 = (const float*)d_in[10]; const float* be1 = (const float*)d_in[11];
    const float* W2 = (const float*)d_in[12]; const float* b2 = (const float*)d_in[13];
    const float* g2 = (const float*)d_in[14]; const float* be2 = (const float*)d_in[15];
    const float* W3 = (const float*)d_in[16]; const float* b3 = (const float*)d_in[17];
    const float* g3 = (const float*)d_in[18]; const float* be3 = (const float*)d_in[19];
    float* out = (float*)d_out;

    char* ws = (char*)d_ws;
    float*    agg    = (float*)(ws + OFF_AGG);
    float*    h      = (float*)(ws + OFF_H);
    float*    score  = (float*)(ws + OFF_SCORE);
    _Float16* flat   = (_Float16*)(ws + OFF_FLAT);
    _Float16* W1h    = (_Float16*)(ws + OFF_W1H);
    _Float16* W2h    = (_Float16*)(ws + OFF_W2H);
    float*    y1     = (float*)(ws + OFF_Y1);
    _Float16* h1h    = (_Float16*)(ws + OFF_H1H);
    float*    y2     = (float*)(ws + OFF_Y2);
    _Float16* h2h    = (_Float16*)(ws + OFF_H2H);
    int*      deg    = (int*)(ws + OFF_DEG);
    int*      rowptr = (int*)(ws + OFF_ROWPTR);
    int*      cursor = (int*)(ws + OFF_CURSOR);
    int*      csrc   = (int*)(ws + OFF_CSRC);

    const int* src = eidx;
    const int* dst = eidx + NEDGE;

    // weight conversions (independent of graph data)
    cvt_f32_f16<<<(HID1 * FIRSTIN + 255) / 256, 256, 0, stream>>>(W1, W1h, HID1 * FIRSTIN);
    cvt_f32_f16<<<(HID2 * HID1 + 255) / 256, 256, 0, stream>>>(W2, W2h, HID2 * HID1);

    // dst-CSR (shared across the 64 identical-topology graphs)
    zero_b32<<<(NNODES + 255) / 256, 256, 0, stream>>>((unsigned*)deg, NNODES);
    hist_dst<<<(NEDGE + 255) / 256, 256, 0, stream>>>(dst, deg);
    scan_rowptr<<<1, 1024, 0, stream>>>(deg, rowptr, cursor);
    fill_csr<<<(NEDGE + 255) / 256, 256, 0, stream>>>(src, dst, cursor, csrc);

    // GraphConv1: agg = gather-sum(x) ; h = relu(conv)  (no atomics, no zero-fill)
    gather_agg<<<(BATCH * NNODES + 255) / 256, 256, 0, stream>>>(data, rowptr, csrc, agg);
    conv1_wmma<<<BATCH * (NNODES / 16), 32, 0, stream>>>(data, agg, W_rel1, W_root1, b_rel1, h);

    // scorer GraphConv: agg = gather-sum(h) ; score = tanh(...)
    gather_agg<<<(BATCH * NNODES + 255) / 256, 256, 0, stream>>>(h, rowptr, csrc, agg);
    scorer<<<(BATCH * NNODES + 255) / 256, 256, 0, stream>>>(agg, h, Wp_rel, bp_rel, Wp_root, score);

    // SAGPool top-K (sorted desc, lowest-index ties) + scaled gather -> flat f16
    topk_gather<<<BATCH, 1024, 16384 * sizeof(unsigned long long), stream>>>(score, h, flat);

    // MLP (GEMM1 split-K x8 for occupancy: 1024 waves over 16 MB of W1)
    zero_b32<<<(BATCH * HID1 + 255) / 256, 256, 0, stream>>>((unsigned*)y1, BATCH * HID1);
    gemm_wmma_splitk<<<(BATCH / 16) * (HID1 / 16) * (FIRSTIN / KCHUNK1), 32, 0, stream>>>(
        flat, W1h, b1, y1, HID1, KCHUNK1);
    bn_relu_f16<<<(HID1 + 255) / 256, 256, 0, stream>>>(y1, g1, be1, h1h, HID1, 1);

    zero_b32<<<(BATCH * HID2 + 255) / 256, 256, 0, stream>>>((unsigned*)y2, BATCH * HID2);
    gemm_wmma_splitk<<<(BATCH / 16) * (HID2 / 16), 32, 0, stream>>>(h1h, W2h, b2, y2, HID2, HID1);
    bn_relu_f16<<<1, HID2, 0, stream>>>(y2, g2, be2, h2h, HID2, 1);

    head<<<1, BATCH, 0, stream>>>(h2h, W3, b3, g3, be3, out);

    (void)in_sizes; (void)n_in; (void)out_size; (void)ws_size;
}
